// PermuInvLoss_27212912788010
// MI455X (gfx1250) — compile-verified, matrix operand order checked by reading, not run on hardware
//
#include <hip/hip_runtime.h>
#include <hip/hip_bf16.h>

typedef int v8i __attribute__((ext_vector_type(8)));

static constexpr int kP   = 4096;       // sampled points per batch
static constexpr int kHW  = 512 * 512;  // spatial size
static constexpr int kBS  = 4;          // batch
static constexpr int kNT  = kP / 16;    // 256 16-wide tiles per side
static constexpr int kJB  = kNT / 8;    // 32 j-strips (8 tiles each) per row

__global__ void zero_out_kernel(float* out) { out[0] = 0.0f; }

__global__ void gather_kernel(const float* __restrict__ y_pred,
                              const int*   __restrict__ y_true,
                              const int*   __restrict__ smpl,
                              float*       __restrict__ pred_s,
                              float*       __restrict__ rp_s,
                              unsigned char* __restrict__ cls_s) {
    int tid = blockIdx.x * blockDim.x + threadIdx.x;   // 0 .. kBS*kP-1
    int b   = tid >> 12;                                // /4096
    int idx = smpl[tid];
    float v = y_pred[b * kHW + idx];
    pred_s[tid] = v;
    rp_s[tid]   = fmaxf(v, 0.0f);
    cls_s[tid]  = (unsigned char)(y_true[b * kHW + idx] & 7);
}

// one-hot B operand (64x16 u8): classes 0..7 live in K=0..7, i.e. lanes 0-15,
// VGPR0 (K=0..3) and VGPR1 (K=4..7); lanes 16-31 hold K>=16 -> zero.
__device__ __forceinline__ v8i make_onehot(unsigned c, int hi) {
    unsigned long long oh = (hi == 0) ? (1ull << (8u * c)) : 0ull;
    v8i V = {0, 0, 0, 0, 0, 0, 0, 0};
    V[0] = (int)(unsigned)(oh & 0xFFFFFFFFull);
    V[1] = (int)(unsigned)(oh >> 32);
    return V;
}

// Consume one 16x16 t-tile: 8 huber pair-losses for this lane's column.
// lim = j_global - i_globalBase: element r contributes iff r < lim (triu).
__device__ __forceinline__ float tile_huber(v8i T, const float rows[8],
                                            float pj, int lim) {
    float a = 0.0f;
#pragma unroll
    for (int r = 0; r < 8; ++r) {
        float pi = fmaxf(rows[r] - pj, 0.0f);     // relu(pred_i - relu(pred_j))
        // 3*huber(pi):   pi<0.1 ? 15*pi^2 : 3*pi-0.15
        float c1 = (pi < 0.1f) ? 15.0f * pi * pi : fmaf(3.0f, pi, -0.15f);
        // huber(1-pi):   y<0.1 ? 5*y^2 : y-0.05   (MARGIN = 1.0)
        float y  = 1.0f - pi;
        float h2 = (y < 0.1f) ? 5.0f * y * y : (y - 0.05f);
        float c  = (T[r] != 0) ? c1 : h2;
        a += (r < lim) ? c : 0.0f;
    }
    return a;
}

// One wave32 per (i-tile x 8 j-tiles) strip. Label-equality t-matrix per
// 16x16 tile via V_WMMA_I32_16X16X64_IU8 on one-hot bytes. The 8 WMMAs are
// software-pipelined: tile u+1's WMMA issues before tile u's result is
// consumed, so the WMMA->VALU hazard window is filled with huber math.
__global__ void pair_loss_kernel(const float* __restrict__ pred_s,
                                 const float* __restrict__ rp_s,
                                 const unsigned char* __restrict__ cls_s,
                                 float* __restrict__ out, float scale) {
    __shared__ float partials[8];

    const int wid  = threadIdx.x >> 5;                  // wave in block (0..7)
    const int wave = blockIdx.x * 8 + wid;
    const int lane = threadIdx.x & 31;

    // decode: batch (x8192) -> i-tile (x32) -> j-strip
    const int b      = wave >> 13;                      // 256*32 waves per batch
    const int rem    = wave & 8191;
    const int ti     = rem >> 5;                        // 0..255
    const int tjBase = (rem & 31) * 8;                  // first j-tile of strip

    float acc = 0.0f;

    if (tjBase + 7 >= ti) {   // strip touches the upper triangle (wave-uniform)
        const int base = b * kP;
        const int i0 = ti * 16;
        const int m  = lane & 15;       // col within tile (C/D: N = lane%16)
        const int hi = lane >> 4;       // lane half      (C/D: M = r + 8*hi)

        // A operand: one-hot of row labels (16x64 u8), built once per strip
        v8i A = make_onehot(cls_s[base + i0 + m], hi);
        v8i C = {0, 0, 0, 0, 0, 0, 0, 0};

        // this lane's 8 row values, loaded once per strip
        const float4* p4 = (const float4*)(pred_s + base + i0 + 8 * hi);
        float4 pa = p4[0], pb = p4[1];
        float rows[8] = {pa.x, pa.y, pa.z, pa.w, pb.x, pb.y, pb.z, pb.w};
        const int ibase = i0 + 8 * hi;                  // global row base

        // prefetch all per-tile operands for the strip (one load clause)
        unsigned cjv[8];
        float    pjv[8];
#pragma unroll
        for (int u = 0; u < 8; ++u) {
            const int j = (tjBase + u) * 16 + m;
            cjv[u] = cls_s[base + j];
            pjv[u] = rp_s[base + j];                    // relu(pred_j)
        }

        // software-pipelined WMMA chain
        v8i Tp = __builtin_amdgcn_wmma_i32_16x16x64_iu8(
            false, A, false, make_onehot(cjv[0], hi), C, false, false);
#pragma unroll
        for (int u = 0; u < 8; ++u) {
            v8i Tn = Tp;
            if (u + 1 < 8) {
                Tn = __builtin_amdgcn_wmma_i32_16x16x64_iu8(
                    false, A, false, make_onehot(cjv[u + 1], hi), C,
                    false, false);
            }
            const int lim = (tjBase + u) * 16 + m - ibase;  // triu cutoff
            acc += tile_huber(Tp, rows, pjv[u], lim);
            Tp = Tn;
        }
    }

    // wave32 reduction -> LDS -> one scaled atomic per block
    for (int off = 16; off > 0; off >>= 1)
        acc += __shfl_xor(acc, off, 32);
    if (lane == 0) partials[wid] = acc;
    __syncthreads();
    if (threadIdx.x == 0) {
        float s = 0.0f;
#pragma unroll
        for (int w = 0; w < 8; ++w) s += partials[w];
        atomicAdd(out, s * scale);
    }
}

extern "C" void kernel_launch(void* const* d_in, const int* in_sizes, int n_in,
                              void* d_out, int out_size, void* d_ws, size_t ws_size,
                              hipStream_t stream) {
    const float* y_pred = (const float*)d_in[0];
    const int*   y_true = (const int*)d_in[1];
    const int*   smpl   = (const int*)d_in[2];
    float* out = (float*)d_out;

    float* pred_s = (float*)d_ws;
    float* rp_s   = pred_s + kBS * kP;
    unsigned char* cls_s = (unsigned char*)(rp_s + kBS * kP);

    zero_out_kernel<<<1, 1, 0, stream>>>(out);

    gather_kernel<<<(kBS * kP) / 256, 256, 0, stream>>>(y_pred, y_true, smpl,
                                                        pred_s, rp_s, cls_s);

    const double npairs = (double)kP * (kP - 1) / 2.0;
    const float scale = (float)(1.0 / (npairs * (double)kBS));
    const int total_waves = kBS * kNT * kJB;   // 4*256*32 = 32768 waves
    pair_loss_kernel<<<total_waves / 8, 256, 0, stream>>>(pred_s, rp_s, cls_s,
                                                          out, scale);
}
